// MambaMILModel_73151882985984
// MI455X (gfx1250) — compile-verified, matrix operand order checked by reading, not run on hardware
//
#include <hip/hip_runtime.h>
#include <hip/hip_bf16.h>
#include <math.h>

// ---------------------------------------------------------------------------
// MambaMIL forward for MI455X (gfx1250, wave32).
// GEMMs: f16 operands (pre-converted activations + pre-transposed weights),
// f32 WMMA accumulation (v_wmma_f32_16x16x32_f16).
//  - A tiles stream via GLOBAL_LOAD_ASYNC_TO_LDS_B128 (ASYNCcnt)
//  - B (weight) tiles stream via the Tensor Data Mover (TENSOR_LOAD_TO_LDS,
//    TENSORcnt), with D# LDS-padding reproducing the bank-conflict-free
//    layout and tensor_dim clamping providing hardware zero-fill at N edges.
// Double-buffered LDS, one barrier per K-step. Scan/layernorm/softmax fp32.
// ---------------------------------------------------------------------------

typedef __attribute__((ext_vector_type(16))) _Float16 v16h;
typedef __attribute__((ext_vector_type(8)))  _Float16 v8h;
typedef __attribute__((ext_vector_type(8)))  float    v8f;
typedef __attribute__((ext_vector_type(4)))  unsigned int v4u;
typedef __attribute__((ext_vector_type(8)))  int      v8i;
typedef __attribute__((ext_vector_type(4)))  int      v4i;

#define SEQ_L   4096
#define DMODEL  512
#define DINNER  1024
#define DSTATE  16
#define DTRANK  64
#define XDBL_N  96
#define ATT_HD  128

#define TM 128
#define TN 128
#define TK 32
#define LPAD 8           // halves of row padding in LDS (16B = 4 DWORDs)

__device__ __forceinline__ float act_apply(float v, int act) {
  switch (act) {
    case 1: return v > 0.0f ? v : 0.0f;                       // relu
    case 2: return tanhf(v);                                  // tanh
    case 3: return (v > 20.0f) ? v : log1pf(__expf(v));       // softplus
    default: return v;
  }
}

// Async 16-byte global -> LDS copy (gfx1250 ASYNCcnt path).
__device__ __forceinline__ void async_copy16(_Float16* lds_dst,
                                             const _Float16* gsrc) {
  unsigned lds_off = (unsigned)(size_t)lds_dst;   // low 32 bits = LDS address
  asm volatile("global_load_async_to_lds_b128 %0, %1, off"
               :: "v"(lds_off), "v"(gsrc)
               : "memory");
}
__device__ __forceinline__ void wait_async0() {
  asm volatile("s_wait_asynccnt 0x0" ::: "memory");
}

// TDM load of a 2D f16 tile [rows x 32 halves] into LDS with 8-half row pad.
// rows clamped by tensor_dim1 (HW zero-fills OOB rows).  One issue per wave.
__device__ __forceinline__ void tdm_load_tile(_Float16* lds_dst,
                                              const _Float16* gsrc,
                                              int row_stride_halves,
                                              int rows_valid) {
  unsigned lds_addr = (unsigned)(size_t)lds_dst;
  unsigned long long ga = (unsigned long long)(size_t)gsrc;
  // ---- D# group 0: count=1 | lds_addr | global_addr | type=2 ----
  v4u g0;
  g0[0] = 1u;                                     // count=1, user descriptor
  g0[1] = lds_addr;
  g0[2] = (unsigned)(ga & 0xFFFFFFFFu);
  g0[3] = (unsigned)((ga >> 32) & 0x1FFFFFFu) | (2u << 30);   // type=2 "image"
  // ---- D# group 1 ----
  // w0: workgroup_mask=0 | data_size=1 (2B) | pad_enable | pad_interval=3
  //     (16 DWORDs = one 32-half row) | pad_amount=3 (4 DWORDs = 8 halves)
  v8i g1;
  g1[0] = (1 << 16) | (1 << 20) | (3 << 22) | (3 << 25);
  g1[1] = (32 & 0xFFFF) << 16;                    // tensor_dim0 = 32 (lo16)
  g1[2] = (32 >> 16) | ((rows_valid & 0xFFFF) << 16);   // td0 hi | td1 lo
  g1[3] = ((rows_valid >> 16) & 0xFFFF) | (32 << 16);   // td1 hi | tile_dim0=32
  g1[4] = TN;                                     // tile_dim1=128, tile_dim2=0
  g1[5] = row_stride_halves;                      // tensor_dim0_stride lo32
  g1[6] = 0;                                      // stride hi | dim1_stride lo
  g1[7] = 0;                                      // dim1_stride hi
  v4i g2 = (v4i){0, 0, 0, 0};
  v4i g3 = (v4i){0, 0, 0, 0};
  v8i g4 = (v8i){0, 0, 0, 0, 0, 0, 0, 0};
  __builtin_amdgcn_tensor_load_to_lds(g0, g1, g2, g3, g4, 0);
}

// ---------------------------------------------------------------------------
// f32 -> f16 elementwise
// ---------------------------------------------------------------------------
__global__ __launch_bounds__(256) void f32_to_f16_kernel(
    const float* __restrict__ in, _Float16* __restrict__ out, int n) {
  int i = blockIdx.x * 256 + threadIdx.x;
  if (i < n) out[i] = (_Float16)in[i];
}

// ---------------------------------------------------------------------------
// f32 [K][N] -> f16 transposed [N][K]
// ---------------------------------------------------------------------------
__global__ __launch_bounds__(256) void f32_to_f16_t_kernel(
    const float* __restrict__ in, _Float16* __restrict__ out, int K, int N) {
  int idx = blockIdx.x * 256 + threadIdx.x;
  if (idx >= N * K) return;
  int n = idx / K;
  int k = idx - n * K;
  out[(size_t)n * K + k] = (_Float16)in[(size_t)k * N + n];
}

// ---------------------------------------------------------------------------
// WMMA GEMM:  C[M,N] = act( A[M,K](f16,row,lda) * Bt[N,K](f16,row,ldb)^T + bias )
// 256 threads (8 waves, wave grid 2x4). Block tile 128x128, K-step 32,
// double-buffered LDS: A via per-lane async-to-LDS, Bt via TDM (wave 0).
// M multiple of 128, K multiple of 32; N edge zero-filled by TDM + store guard.
// ---------------------------------------------------------------------------
__global__ __launch_bounds__(256) void wmma_gemm_kernel(
    const _Float16* __restrict__ A, int lda,
    const _Float16* __restrict__ Bt, int ldb,
    float* __restrict__ C, int ldc,
    _Float16* __restrict__ C16,
    const float* __restrict__ bias,
    int M, int N, int K, int act) {
  __shared__ __align__(16) _Float16 As[2][TM][TK + LPAD];
  __shared__ __align__(16) _Float16 Bs[2][TN][TK + LPAD];

  const int tid  = threadIdx.x;
  const int wave = tid >> 5;
  const int lane = tid & 31;
  const int wm   = wave >> 2;           // 0..1 : 64-row strip
  const int wn   = wave & 3;            // 0..3 : 32-col strip
  const int rowBase = blockIdx.y * TM;
  const int colBase = blockIdx.x * TN;
  const int m16  = lane & 15;
  const int kb   = (lane < 16) ? 0 : 8; // documented 16-bit fragment layout
  const int rOff = (lane < 16) ? 0 : 8;
  const int rowsValid = N - colBase;    // TDM zero-fills rows beyond this

  v8f acc[4][2];
  #pragma unroll
  for (int mt = 0; mt < 4; ++mt)
    #pragma unroll
    for (int nt = 0; nt < 2; ++nt) acc[mt][nt] = (v8f){0,0,0,0,0,0,0,0};

  // stage A tile: 512 x 16B async copies, 2 per thread, branch-free
  auto stageA = [&](int k0, int b) {
    #pragma unroll
    for (int i = 0; i < 2; ++i) {
      int e = tid + i * 256;                 // 0..511
      int r = e >> 2, c = (e & 3) * 8;
      async_copy16(&As[b][r][c], A + (size_t)(rowBase + r) * lda + k0 + c);
    }
  };
  // stage B tile: one TDM descriptor, issued by wave 0 only
  auto stageB = [&](int k0, int b) {
    if (wave == 0)
      tdm_load_tile(&Bs[b][0][0], Bt + (size_t)colBase * ldb + k0,
                    ldb, rowsValid);
  };

  const int nk = K / TK;
  stageA(0, 0);
  stageB(0, 0);

  for (int ks = 0; ks < nk; ++ks) {
    wait_async0();
    __builtin_amdgcn_s_wait_tensorcnt(0);
    __syncthreads();                          // tile ks visible to all waves
    if (ks + 1 < nk) {                        // overlap next tile with compute
      stageA((ks + 1) * TK, (ks + 1) & 1);
      stageB((ks + 1) * TK, (ks + 1) & 1);
    }

    const int b = ks & 1;
    v16h af[4];
    #pragma unroll
    for (int mt = 0; mt < 4; ++mt) {
      v8h lo = *(const v8h*)&As[b][wm * 64 + mt * 16 + m16][kb];
      v8h hi = *(const v8h*)&As[b][wm * 64 + mt * 16 + m16][kb + 16];
      #pragma unroll
      for (int i = 0; i < 8; ++i) { af[mt][i] = lo[i]; af[mt][8 + i] = hi[i]; }
    }
    #pragma unroll
    for (int nt = 0; nt < 2; ++nt) {
      v8h blo = *(const v8h*)&Bs[b][wn * 32 + nt * 16 + m16][kb];
      v8h bhi = *(const v8h*)&Bs[b][wn * 32 + nt * 16 + m16][kb + 16];
      v16h bf;
      #pragma unroll
      for (int i = 0; i < 8; ++i) { bf[i] = blo[i]; bf[8 + i] = bhi[i]; }
      #pragma unroll
      for (int mt = 0; mt < 4; ++mt)
        acc[mt][nt] = __builtin_amdgcn_wmma_f32_16x16x32_f16(
            false, af[mt], false, bf, (short)0, acc[mt][nt], false, false);
    }
  }

  // epilogue: D layout — vgpr r: M = r + 8*(lane>=16), N = lane&15
  #pragma unroll
  for (int mt = 0; mt < 4; ++mt) {
    #pragma unroll
    for (int nt = 0; nt < 2; ++nt) {
      int col = colBase + wn * 32 + nt * 16 + m16;
      if (col >= N) continue;
      float bv = bias ? bias[col] : 0.0f;
      #pragma unroll
      for (int r = 0; r < 8; ++r) {
        int row = rowBase + wm * 64 + mt * 16 + r + rOff;
        float v = act_apply(acc[mt][nt][r] + bv, act);
        C[(size_t)row * ldc + col] = v;
        if (C16) C16[(size_t)row * ldc + col] = (_Float16)v;
      }
    }
  }
}

// ---------------------------------------------------------------------------
// LayerNorm over last dim D, one block per row, optional residual, optional
// f16 mirror output (feeds the next WMMA GEMM).
// ---------------------------------------------------------------------------
__global__ __launch_bounds__(256) void layernorm_kernel(
    const float* __restrict__ in, const float* __restrict__ res,
    const float* __restrict__ w, const float* __restrict__ b,
    float* __restrict__ out, _Float16* __restrict__ out16, int D) {
  __shared__ float red[256];
  const int row = blockIdx.x;
  const int tid = threadIdx.x;
  const float* x = in + (size_t)row * D;
  const float* r = res ? res + (size_t)row * D : nullptr;

  float s = 0.0f;
  for (int i = tid; i < D; i += 256) s += x[i] + (r ? r[i] : 0.0f);
  red[tid] = s; __syncthreads();
  for (int o = 128; o > 0; o >>= 1) { if (tid < o) red[tid] += red[tid + o]; __syncthreads(); }
  float mean = red[0] / (float)D;
  __syncthreads();

  float vs = 0.0f;
  for (int i = tid; i < D; i += 256) {
    float v = x[i] + (r ? r[i] : 0.0f) - mean;
    vs += v * v;
  }
  red[tid] = vs; __syncthreads();
  for (int o = 128; o > 0; o >>= 1) { if (tid < o) red[tid] += red[tid + o]; __syncthreads(); }
  float inv = rsqrtf(red[0] / (float)D + 1e-5f);

  for (int i = tid; i < D; i += 256) {
    float v = (x[i] + (r ? r[i] : 0.0f) - mean) * inv * w[i] + b[i];
    out[(size_t)row * D + i] = v;
    if (out16) out16[(size_t)row * D + i] = (_Float16)v;
  }
}

// ---------------------------------------------------------------------------
// Depthwise causal conv (D_CONV=4) + SiLU; writes f32 (for scan) + f16 (GEMM).
// ---------------------------------------------------------------------------
__global__ __launch_bounds__(256) void conv_silu_kernel(
    const float* __restrict__ xz,       // L x 2*DINNER
    const float* __restrict__ Wc,       // DINNER x 4
    const float* __restrict__ bc,       // DINNER
    float* __restrict__ xc, _Float16* __restrict__ xc16) {
  int idx = blockIdx.x * 256 + threadIdx.x;
  if (idx >= SEQ_L * DINNER) return;
  int l = idx / DINNER;
  int d = idx - l * DINNER;
  float acc = 0.0f;
  #pragma unroll
  for (int k = 0; k < 4; ++k) {
    int lp = l - 3 + k;
    if (lp >= 0) acc += Wc[d * 4 + k] * xz[(size_t)lp * (2 * DINNER) + d];
  }
  acc += bc[d];
  float v = acc / (1.0f + __expf(-acc));   // silu
  xc[idx] = v;
  xc16[idx] = (_Float16)v;
}

// ---------------------------------------------------------------------------
// Selective scan. Lane = (d, s): 16384 lanes = 512 waves. Cross-lane sum over
// 16 states via shfl_xor(width=16). Fuses +x*D and *silu(z); writes f16 y.
// ---------------------------------------------------------------------------
__global__ __launch_bounds__(256) void scan_kernel(
    const float* __restrict__ xc,      // L x DINNER
    const float* __restrict__ delta,   // L x DINNER
    const float* __restrict__ xdbl,    // L x 96 (dt | B | C)
    const float* __restrict__ xz,      // L x 2*DINNER (z at +DINNER)
    const float* __restrict__ A_log,   // DINNER x DSTATE (layer slice)
    const float* __restrict__ Dsk,     // DINNER
    _Float16* __restrict__ y16) {      // L x DINNER
  int t = blockIdx.x * 256 + threadIdx.x;   // 0..16383
  int d = t >> 4;
  int s = t & 15;
  float Ac = -__expf(A_log[d * DSTATE + s]);
  float Dd = Dsk[d];
  float state = 0.0f;
  for (int l = 0; l < SEQ_L; ++l) {
    float dl = delta[(size_t)l * DINNER + d];
    float xv = xc[(size_t)l * DINNER + d];
    float Bv = xdbl[(size_t)l * XDBL_N + DTRANK + s];
    float Cv = xdbl[(size_t)l * XDBL_N + DTRANK + DSTATE + s];
    state = __expf(dl * Ac) * state + dl * Bv * xv;
    float part = state * Cv;
    part += __shfl_xor(part, 1, 16);
    part += __shfl_xor(part, 2, 16);
    part += __shfl_xor(part, 4, 16);
    part += __shfl_xor(part, 8, 16);
    if (s == 0) {
      float z = xz[(size_t)l * (2 * DINNER) + DINNER + d];
      float sz = z / (1.0f + __expf(-z));
      y16[(size_t)l * DINNER + d] = (_Float16)((part + xv * Dd) * sz);
    }
  }
}

// ---------------------------------------------------------------------------
// scores[l] = t[l,:] . Wa2 + ba2   (one wave per row)
// ---------------------------------------------------------------------------
__global__ __launch_bounds__(256) void score_kernel(
    const float* __restrict__ t, const float* __restrict__ Wa2,
    const float* __restrict__ ba2, float* __restrict__ sc) {
  int row  = (blockIdx.x * 256 + threadIdx.x) >> 5;
  int lane = threadIdx.x & 31;
  if (row >= SEQ_L) return;
  float sum = 0.0f;
  for (int i = lane; i < ATT_HD; i += 32) sum += t[(size_t)row * ATT_HD + i] * Wa2[i];
  sum += __shfl_xor(sum, 1,  32);
  sum += __shfl_xor(sum, 2,  32);
  sum += __shfl_xor(sum, 4,  32);
  sum += __shfl_xor(sum, 8,  32);
  sum += __shfl_xor(sum, 16, 32);
  if (lane == 0) sc[row] = sum + ba2[0];
}

// ---------------------------------------------------------------------------
// In-place softmax over L (single block).
// ---------------------------------------------------------------------------
__global__ __launch_bounds__(256) void softmax_kernel(float* __restrict__ sc, int n) {
  __shared__ float red[256];
  const int tid = threadIdx.x;
  float m = -1e30f;
  for (int i = tid; i < n; i += 256) m = fmaxf(m, sc[i]);
  red[tid] = m; __syncthreads();
  for (int o = 128; o > 0; o >>= 1) { if (tid < o) red[tid] = fmaxf(red[tid], red[tid + o]); __syncthreads(); }
  m = red[0]; __syncthreads();
  float s = 0.0f;
  for (int i = tid; i < n; i += 256) s += __expf(sc[i] - m);
  red[tid] = s; __syncthreads();
  for (int o = 128; o > 0; o >>= 1) { if (tid < o) red[tid] += red[tid + o]; __syncthreads(); }
  float inv = 1.0f / red[0]; __syncthreads();
  for (int i = tid; i < n; i += 256) sc[i] = __expf(sc[i] - m) * inv;
}

// ---------------------------------------------------------------------------
// pooled[d] = sum_l aw[l] * h[l,d]
// ---------------------------------------------------------------------------
__global__ __launch_bounds__(256) void pool_kernel(
    const float* __restrict__ aw, const float* __restrict__ h,
    float* __restrict__ pooled) {
  int d = blockIdx.x * 256 + threadIdx.x;
  if (d >= DMODEL) return;
  float acc = 0.0f;
  for (int l = 0; l < SEQ_L; ++l) acc += aw[l] * h[(size_t)l * DMODEL + d];
  pooled[d] = acc;
}

// ---------------------------------------------------------------------------
// out[j] = pooled . Wh[:,j] + bh[j]   (HEAD = 2)
// ---------------------------------------------------------------------------
__global__ __launch_bounds__(256) void head_kernel(
    const float* __restrict__ pooled, const float* __restrict__ Wh,
    const float* __restrict__ bh, float* __restrict__ out) {
  __shared__ float r0[256], r1[256];
  const int tid = threadIdx.x;
  float a = 0.0f, b = 0.0f;
  for (int d = tid; d < DMODEL; d += 256) {
    float p = pooled[d];
    a += p * Wh[d * 2 + 0];
    b += p * Wh[d * 2 + 1];
  }
  r0[tid] = a; r1[tid] = b; __syncthreads();
  for (int o = 128; o > 0; o >>= 1) {
    if (tid < o) { r0[tid] += r0[tid + o]; r1[tid] += r1[tid + o]; }
    __syncthreads();
  }
  if (tid == 0) { out[0] = r0[0] + bh[0]; out[1] = r1[0] + bh[1]; }
}

// ---------------------------------------------------------------------------
// Host launcher
// ---------------------------------------------------------------------------
extern "C" void kernel_launch(void* const* d_in, const int* in_sizes, int n_in,
                              void* d_out, int out_size, void* d_ws, size_t ws_size,
                              hipStream_t stream) {
  (void)in_sizes; (void)n_in; (void)out_size; (void)ws_size;
  const float* h_in   = (const float*)d_in[0];
  const float* W_proj = (const float*)d_in[1];
  const float* b_proj = (const float*)d_in[2];
  const float* ln_w   = (const float*)d_in[3];
  const float* ln_b   = (const float*)d_in[4];
  const float* W_in   = (const float*)d_in[5];
  const float* W_conv = (const float*)d_in[6];
  const float* b_conv = (const float*)d_in[7];
  const float* W_x    = (const float*)d_in[8];
  const float* W_dt   = (const float*)d_in[9];
  const float* b_dt   = (const float*)d_in[10];
  const float* A_log  = (const float*)d_in[11];
  const float* Dsk    = (const float*)d_in[12];
  const float* W_out  = (const float*)d_in[13];
  const float* norm_w = (const float*)d_in[14];
  const float* norm_b = (const float*)d_in[15];
  const float* Wa1    = (const float*)d_in[16];
  const float* ba1    = (const float*)d_in[17];
  const float* Wa2    = (const float*)d_in[18];
  const float* ba2    = (const float*)d_in[19];
  const float* Wh     = (const float*)d_in[20];
  const float* bh     = (const float*)d_in[21];
  float* out = (float*)d_out;

  // ---- workspace carve-up ----
  char* ws = (char*)d_ws;
  size_t off = 0;
  auto take = [&](size_t bytes) -> char* {
    char* p = ws + off;
    off = (off + bytes + 255) & ~(size_t)255;
    return p;
  };
  // f32 activations
  float* h0     = (float*)take((size_t)SEQ_L * DMODEL * 4);
  float* hcur   = (float*)take((size_t)SEQ_L * DMODEL * 4);
  float* lnbuf  = (float*)take((size_t)SEQ_L * DMODEL * 4);
  float* xz     = (float*)take((size_t)SEQ_L * 2 * DINNER * 4);
  float* xc     = (float*)take((size_t)SEQ_L * DINNER * 4);
  float* xdbl   = (float*)take((size_t)SEQ_L * XDBL_N * 4);
  float* dbuf   = (float*)take((size_t)SEQ_L * DINNER * 4);
  float* tatt   = (float*)take((size_t)SEQ_L * ATT_HD * 4);
  float* sc     = (float*)take((size_t)SEQ_L * 4);
  float* pooled = (float*)take((size_t)DMODEL * 4);
  // f16 activations (GEMM A operands)
  _Float16* hin16  = (_Float16*)take((size_t)SEQ_L * 1024 * 2);
  _Float16* ln16   = (_Float16*)take((size_t)SEQ_L * DMODEL * 2);
  _Float16* xc16   = (_Float16*)take((size_t)SEQ_L * DINNER * 2);
  _Float16* xdbl16 = (_Float16*)take((size_t)SEQ_L * XDBL_N * 2);
  _Float16* y16    = (_Float16*)take((size_t)SEQ_L * DINNER * 2);
  // f16 transposed weights [N][K]
  _Float16* wpT   = (_Float16*)take((size_t)512 * 1024 * 2);
  _Float16* winT  = (_Float16*)take((size_t)2 * 2048 * 512 * 2);
  _Float16* wxT   = (_Float16*)take((size_t)2 * 96 * 1024 * 2);
  _Float16* wdtT  = (_Float16*)take((size_t)2 * 1024 * 64 * 2);
  _Float16* woutT = (_Float16*)take((size_t)2 * 512 * 1024 * 2);
  _Float16* wa1T  = (_Float16*)take((size_t)128 * 512 * 2);
  (void)take(65536);   // tail pad

  auto cvt = [&](const float* src, _Float16* dst, int n) {
    f32_to_f16_kernel<<<(n + 255) / 256, 256, 0, stream>>>(src, dst, n);
  };
  auto cvtT = [&](const float* src, _Float16* dst, int K, int N) {
    f32_to_f16_t_kernel<<<(N * K + 255) / 256, 256, 0, stream>>>(src, dst, K, N);
  };
  auto gemm = [&](const _Float16* A, int lda, const _Float16* Bt, int ldb,
                  float* C, int ldc, _Float16* C16, const float* bias,
                  int M, int N, int K, int act) {
    dim3 g((N + TN - 1) / TN, (M + TM - 1) / TM);
    wmma_gemm_kernel<<<g, 256, 0, stream>>>(A, lda, Bt, ldb, C, ldc, C16, bias,
                                            M, N, K, act);
  };

  // ---- prepare f16 operands ----
  cvt(h_in, hin16, SEQ_L * 1024);
  cvtT(W_proj, wpT, 1024, 512);
  for (int l = 0; l < 2; ++l) {
    cvtT(W_in  + (size_t)l * 512 * 2048, winT  + (size_t)l * 2048 * 512, 512, 2048);
    cvtT(W_x   + (size_t)l * 1024 * 96,  wxT   + (size_t)l * 96 * 1024,  1024, 96);
    cvtT(W_dt  + (size_t)l * 64 * 1024,  wdtT  + (size_t)l * 1024 * 64,  64, 1024);
    cvtT(W_out + (size_t)l * 1024 * 512, woutT + (size_t)l * 512 * 1024, 1024, 512);
  }
  cvtT(Wa1, wa1T, 512, 128);

  // ---- input projection: relu(h @ W_proj + b_proj) -> h0 ----
  gemm(hin16, 1024, wpT, 1024, h0, DMODEL, nullptr, b_proj,
       SEQ_L, DMODEL, 1024, /*relu*/1);

  // ---- Mamba layers ----
  for (int l = 0; l < 2; ++l) {
    const float* hsrc = (l == 0) ? h0 : hcur;
    layernorm_kernel<<<SEQ_L, 256, 0, stream>>>(
        hsrc, nullptr, ln_w + l * DMODEL, ln_b + l * DMODEL, lnbuf, ln16, DMODEL);

    // xz = ln_h @ W_in[l]        (4096x512 x 512x2048)
    gemm(ln16, DMODEL, winT + (size_t)l * 2048 * 512, DMODEL,
         xz, 2 * DINNER, nullptr, nullptr, SEQ_L, 2 * DINNER, DMODEL, 0);

    // depthwise conv + silu on x half
    conv_silu_kernel<<<(SEQ_L * DINNER + 255) / 256, 256, 0, stream>>>(
        xz, W_conv + (size_t)l * DINNER * 4, b_conv + (size_t)l * DINNER, xc, xc16);

    // x_dbl = x @ W_x[l]         (4096x1024 x 1024x96), f16 mirror for dt GEMM
    gemm(xc16, DINNER, wxT + (size_t)l * 96 * 1024, DINNER,
         xdbl, XDBL_N, xdbl16, nullptr, SEQ_L, XDBL_N, DINNER, 0);

    // delta = softplus(dt @ W_dt[l] + b_dt)   (A = x_dbl16[:, :64], lda=96)
    gemm(xdbl16, XDBL_N, wdtT + (size_t)l * 1024 * 64, DTRANK,
         dbuf, DINNER, nullptr, b_dt + (size_t)l * DINNER,
         SEQ_L, DINNER, DTRANK, /*softplus*/3);

    // selective scan -> y16 (fused * silu(z))
    scan_kernel<<<(DINNER * DSTATE) / 256, 256, 0, stream>>>(
        xc, dbuf, xdbl, xz,
        A_log + (size_t)l * DINNER * DSTATE, Dsk + (size_t)l * DINNER, y16);

    // h = y @ W_out[l]           (4096x1024 x 1024x512)
    gemm(y16, DINNER, woutT + (size_t)l * 512 * 1024, DINNER,
         hcur, DMODEL, nullptr, nullptr, SEQ_L, DMODEL, DINNER, 0);
  }

  // ---- residual + final layernorm -> lnbuf (f32) / ln16 ----
  layernorm_kernel<<<SEQ_L, 256, 0, stream>>>(hcur, h0, norm_w, norm_b,
                                              lnbuf, ln16, DMODEL);

  // ---- attention pooling ----
  gemm(ln16, DMODEL, wa1T, DMODEL, tatt, ATT_HD, nullptr, ba1,
       SEQ_L, ATT_HD, DMODEL, /*tanh*/2);
  score_kernel<<<(SEQ_L * 32 + 255) / 256, 256, 0, stream>>>(tatt, Wa2, ba2, sc);
  softmax_kernel<<<1, 256, 0, stream>>>(sc, SEQ_L);
  pool_kernel<<<(DMODEL + 255) / 256, 256, 0, stream>>>(sc, lnbuf, pooled);
  head_kernel<<<1, 256, 0, stream>>>(pooled, Wh, bh, out);
}
